// FraudGraphSAGE_87789131530773
// MI455X (gfx1250) — matrix operand order repaired and checked by reading, compiled, not run on hardware
//
#include <hip/hip_runtime.h>
#include <hip/hip_bf16.h>

// ---------------------------------------------------------------------------
// GraphSAGE (2x SAGEConv + linear head) for MI455X / gfx1250 (wave32, WMMA).
//
// Cost model: edge scatter/gather dominates (~1.8 GB of mostly L2-resident
// traffic; agg buffers fit the 192MB L2). Node-level GEMMs (~2.5 GFLOP) use
// v_wmma_f32_16x16x32_f16, one wave per 16-node tile, f32 accumulation, bias
// preloaded into C, fused ReLU. Weight B-fragments are pre-swizzled into the
// WMMA per-lane layout so the hot kernels do only unconditional coalesced
// vector loads (no lane-divergent zero-pad predicates -> no exec-mask chains).
// ---------------------------------------------------------------------------

#define N_NODES 200000
#define N_EDGES 3200000
#define IN_CH   12
#define IN_PAD  16
#define HID     64
#define HID2    32
#define OUT_CH  2

typedef __attribute__((ext_vector_type(16))) _Float16 v16h;
typedef __attribute__((ext_vector_type(8)))  float    v8f;

// ---------------- zero workspace (deg | agg1p | agg2: atomic targets) -------
__global__ __launch_bounds__(256)
void zero_f4_kernel(float4* __restrict__ p, long long n4) {
    long long i = (long long)blockIdx.x * blockDim.x + threadIdx.x;
    long long stride = (long long)gridDim.x * blockDim.x;
    float4 z; z.x = 0.f; z.y = 0.f; z.z = 0.f; z.w = 0.f;
    for (; i < n4; i += stride) p[i] = z;
}

// ---------------- pad x [N,12] -> x_pad [N,16] (tail zeros) -----------------
__global__ __launch_bounds__(256)
void pad_x_kernel(const float* __restrict__ x, float* __restrict__ x_pad) {
    int t = blockIdx.x * blockDim.x + threadIdx.x;   // one thread per (n, c)
    if (t >= N_NODES * IN_PAD) return;
    int n = t >> 4, c = t & 15;
    x_pad[t] = (c < IN_CH) ? x[n * IN_CH + c] : 0.0f;
}

// ---------------- pre-swizzle weights into WMMA B-fragment layout -----------
// B (32x16 f16): lanes 0-15 hold K{0..7,16..23} of column N=lane&15, lanes 16-31
// hold K{8..15,24..31}.  wpack index map:
//   [  0..127]  w1l: frag = tile*32 + lane          (4 tiles, K pad 12->32)
//   [128..255]  w1r: same
//   [256..383]  w2l: frag = (tile*2 + chunk)*32 + lane  (2 tiles x 2 K-chunks)
//   [384..511]  w2r: same
__global__ __launch_bounds__(256)
void pack_weights_kernel(const float* __restrict__ w1l, const float* __restrict__ w1r,
                         const float* __restrict__ w2l, const float* __restrict__ w2r,
                         v16h* __restrict__ wpack) {
    int t = blockIdx.x * blockDim.x + threadIdx.x;
    if (t >= 512) return;
    int group = t >> 7;         // 0:w1l 1:w1r 2:w2l 3:w2r
    int idx   = t & 127;
    int lane  = idx & 31;
    int hi    = (lane >> 4) & 1;
    int col   = lane & 15;
    v16h f;
    if (group < 2) {
        const float* w = (group == 0) ? w1l : w1r;   // [12,64]
        int tile = idx >> 5;                         // 0..3
        int ncol = tile * 16 + col;
        for (int j = 0; j < 16; ++j) {
            int k = ((j >= 8) ? 16 : 0) + hi * 8 + (j & 7);
            f[j] = (_Float16)((k < IN_CH) ? w[k * HID + ncol] : 0.0f);
        }
    } else {
        const float* w = (group == 2) ? w2l : w2r;   // [64,32]
        int tc   = idx >> 5;                         // tile*2 + chunk
        int tile = tc >> 1, chunk = tc & 1;
        int ncol = tile * 16 + col;
        for (int j = 0; j < 16; ++j) {
            int k = chunk * 32 + ((j >= 8) ? 16 : 0) + hi * 8 + (j & 7);
            f[j] = (_Float16)w[k * HID2 + ncol];
        }
    }
    wpack[t] = f;
}

// ---------------- layer-1 scatter: deg count + sum of x into agg1p ----------
__global__ __launch_bounds__(256)
void scatter1_kernel(const float* __restrict__ x,
                     const int* __restrict__ src, const int* __restrict__ dst,
                     float* __restrict__ agg1p, float* __restrict__ deg) {
    int e = blockIdx.x * blockDim.x + threadIdx.x;
    if (e >= N_EDGES) return;
    int s = src[e], d = dst[e];
    atomicAdd(&deg[d], 1.0f);
    const float* xs = x + (size_t)s * IN_CH;
    float* ad = agg1p + (size_t)d * IN_PAD;          // padded row, tail stays 0
#pragma unroll
    for (int c = 0; c < IN_CH; ++c) atomicAdd(&ad[c], xs[c]);
}

// ---------------- layer-1 WMMA: h1 = relu(mean@w1l + b1 + x@w1r) ------------
// One wave -> 16 nodes x 64 outputs. K=12 zero-padded to 32 (pad baked into
// x_pad/agg1p rows and the packed weight fragments: all loads unconditional).
__global__ __launch_bounds__(128)
void layer1_wmma_kernel(const float* __restrict__ x_pad,   // [N,16]
                        const float* __restrict__ agg1p,   // [N,16]
                        const float* __restrict__ deg,
                        const v16h*  __restrict__ wpack,
                        const float* __restrict__ b1,      // [64]
                        float* __restrict__ h1) {          // [N,64]
    int wave = (blockIdx.x * blockDim.x + threadIdx.x) >> 5;
    int lane = threadIdx.x & 31;
    int base = wave * 16;
    if (base >= N_NODES) return;               // wave-uniform (EXEC stays full)

    int row = lane & 15;
    int hi  = (lane >> 4) & 1;
    int node = base + row;
    float rdeg = 1.0f / fmaxf(deg[node], 1.0f);

    const float* arow = agg1p + (size_t)node * IN_PAD;
    const float* xr   = x_pad + (size_t)node * IN_PAD;

    // A (16x32 f16): lanes 0-15 hold K{0..7,16..23} of row M=lane&15,
    // lanes 16-31 hold K{8..15,24..31}.  K>=16 is compile-time zero here.
    v16h a_mean, a_x;
#pragma unroll
    for (int j = 0; j < 8; ++j) {
        a_mean[j] = (_Float16)(arow[hi * 8 + j] * rdeg);
        a_x[j]    = (_Float16)(xr[hi * 8 + j]);
    }
#pragma unroll
    for (int j = 8; j < 16; ++j) {
        a_mean[j] = (_Float16)0.0f;
        a_x[j]    = (_Float16)0.0f;
    }

#pragma unroll
    for (int nt = 0; nt < HID / 16; ++nt) {
        int ncol = nt * 16 + row;
        v16h b_l = wpack[nt * 32 + lane];          // coalesced 32B/lane
        v16h b_r = wpack[128 + nt * 32 + lane];
        v8f acc;
        float bias = b1[ncol];
#pragma unroll
        for (int i = 0; i < 8; ++i) acc[i] = bias;
        acc = __builtin_amdgcn_wmma_f32_16x16x32_f16(false, a_mean, false, b_l,
                                                     (short)0, acc, false, false);
        acc = __builtin_amdgcn_wmma_f32_16x16x32_f16(false, a_x, false, b_r,
                                                     (short)0, acc, false, false);
        // C/D: M = i + hi*8, N = ncol
#pragma unroll
        for (int i = 0; i < 8; ++i) {
            int m = i + hi * 8;
            h1[(size_t)(base + m) * HID + ncol] = fmaxf(acc[i], 0.0f);
        }
    }
}

// ---------------- layer-2 scatter: agg2[dst] += h1[src] (64 lanes/edge) -----
__global__ __launch_bounds__(256)
void scatter2_kernel(const float* __restrict__ h1,
                     const int* __restrict__ src, const int* __restrict__ dst,
                     float* __restrict__ agg2) {
    long long t = (long long)blockIdx.x * blockDim.x + threadIdx.x;
    int e = (int)(t >> 6);
    int c = (int)(t & 63);
    if (e >= N_EDGES) return;
    int s = src[e], d = dst[e];
    atomicAdd(&agg2[(size_t)d * HID + c], h1[(size_t)s * HID + c]);
}

// ---------------- layer-2 WMMA: h2 = relu(mean2@w2l + b2 + h1@w2r) ----------
// K=64 -> two K=32 chunks; N=32 -> two 16-col tiles; 4 WMMAs per tile.
__global__ __launch_bounds__(128)
void layer2_wmma_kernel(const float* __restrict__ h1,    // [N,64]
                        const float* __restrict__ agg2,  // [N,64]
                        const float* __restrict__ deg,
                        const v16h*  __restrict__ wpack,
                        const float* __restrict__ b2,    // [32]
                        float* __restrict__ h2) {        // [N,32]
    int wave = (blockIdx.x * blockDim.x + threadIdx.x) >> 5;
    int lane = threadIdx.x & 31;
    int base = wave * 16;
    if (base >= N_NODES) return;

    int row = lane & 15;
    int hi  = (lane >> 4) & 1;
    int node = base + row;
    float rdeg = 1.0f / fmaxf(deg[node], 1.0f);

    const float* aggrow = agg2 + (size_t)node * HID;
    const float* hrow   = h1   + (size_t)node * HID;

    v16h aM[2], aX[2];
#pragma unroll
    for (int c = 0; c < 2; ++c)
#pragma unroll
        for (int j = 0; j < 16; ++j) {
            int k = c * 32 + ((j >= 8) ? 16 : 0) + hi * 8 + (j & 7);
            aM[c][j] = (_Float16)(aggrow[k] * rdeg);
            aX[c][j] = (_Float16)(hrow[k]);
        }

#pragma unroll
    for (int nt = 0; nt < HID2 / 16; ++nt) {
        int ncol = nt * 16 + row;
        v16h bl0 = wpack[256 + (nt * 2 + 0) * 32 + lane];
        v16h bl1 = wpack[256 + (nt * 2 + 1) * 32 + lane];
        v16h br0 = wpack[384 + (nt * 2 + 0) * 32 + lane];
        v16h br1 = wpack[384 + (nt * 2 + 1) * 32 + lane];
        v8f acc;
        float bias = b2[ncol];
#pragma unroll
        for (int i = 0; i < 8; ++i) acc[i] = bias;
        acc = __builtin_amdgcn_wmma_f32_16x16x32_f16(false, aM[0], false, bl0,
                                                     (short)0, acc, false, false);
        acc = __builtin_amdgcn_wmma_f32_16x16x32_f16(false, aM[1], false, bl1,
                                                     (short)0, acc, false, false);
        acc = __builtin_amdgcn_wmma_f32_16x16x32_f16(false, aX[0], false, br0,
                                                     (short)0, acc, false, false);
        acc = __builtin_amdgcn_wmma_f32_16x16x32_f16(false, aX[1], false, br1,
                                                     (short)0, acc, false, false);
#pragma unroll
        for (int i = 0; i < 8; ++i) {
            int m = i + hi * 8;
            h2[(size_t)(base + m) * HID2 + ncol] = fmaxf(acc[i], 0.0f);
        }
    }
}

// ---------------- classifier: out = h2 @ wc + bc  (32->2, scalar) -----------
__global__ __launch_bounds__(256)
void classifier_kernel(const float* __restrict__ h2,
                       const float* __restrict__ wc,   // [32,2]
                       const float* __restrict__ bc,   // [2]
                       float* __restrict__ out) {      // [N,2]
    int node = blockIdx.x * blockDim.x + threadIdx.x;
    if (node >= N_NODES) return;
    const float* r = h2 + (size_t)node * HID2;
    float o0 = bc[0], o1 = bc[1];
#pragma unroll
    for (int k = 0; k < HID2; ++k) {
        float v = r[k];
        o0 = fmaf(v, wc[k * 2 + 0], o0);
        o1 = fmaf(v, wc[k * 2 + 1], o1);
    }
    out[(size_t)node * 2 + 0] = o0;
    out[(size_t)node * 2 + 1] = o1;
}

extern "C" void kernel_launch(void* const* d_in, const int* in_sizes, int n_in,
                              void* d_out, int out_size, void* d_ws, size_t ws_size,
                              hipStream_t stream) {
    const float* x   = (const float*)d_in[0];
    const int*   ei  = (const int*)d_in[1];      // [2, E] flat
    const float* w1l = (const float*)d_in[2];
    const float* b1  = (const float*)d_in[3];
    const float* w1r = (const float*)d_in[4];
    const float* w2l = (const float*)d_in[5];
    const float* b2  = (const float*)d_in[6];
    const float* w2r = (const float*)d_in[7];
    const float* wc  = (const float*)d_in[8];
    const float* bc  = (const float*)d_in[9];
    float* out = (float*)d_out;

    const int* src = ei;
    const int* dst = ei + N_EDGES;

    // Workspace layout (atomic-accumulation region first, zeroed every launch;
    // all region sizes are multiples of 256B so everything stays aligned).
    char* ws = (char*)d_ws;
    size_t deg_bytes   = (size_t)N_NODES * sizeof(float);              //   0.8 MB
    size_t agg1_bytes  = (size_t)N_NODES * IN_PAD * sizeof(float);     //  12.8 MB
    size_t agg2_bytes  = (size_t)N_NODES * HID * sizeof(float);        //  51.2 MB
    size_t h1_bytes    = (size_t)N_NODES * HID * sizeof(float);        //  51.2 MB
    size_t h2_bytes    = (size_t)N_NODES * HID2 * sizeof(float);       //  25.6 MB
    size_t xpad_bytes  = (size_t)N_NODES * IN_PAD * sizeof(float);     //  12.8 MB

    float* deg   = (float*)(ws);
    float* agg1p = (float*)(ws + deg_bytes);
    float* agg2  = (float*)(ws + deg_bytes + agg1_bytes);
    float* h1    = (float*)(ws + deg_bytes + agg1_bytes + agg2_bytes);
    float* h2    = (float*)(ws + deg_bytes + agg1_bytes + agg2_bytes + h1_bytes);
    float* x_pad = (float*)(ws + deg_bytes + agg1_bytes + agg2_bytes + h1_bytes + h2_bytes);
    v16h*  wpack = (v16h*) (ws + deg_bytes + agg1_bytes + agg2_bytes + h1_bytes + h2_bytes + xpad_bytes);

    // 1) zero atomic-accumulation buffers (deg|agg1p|agg2 contiguous)
    long long zero_n4 = (long long)(deg_bytes + agg1_bytes + agg2_bytes) / 16;
    zero_f4_kernel<<<4096, 256, 0, stream>>>((float4*)ws, zero_n4);

    // 2) per-launch prep: padded x copy + pre-swizzled f16 weight fragments
    pad_x_kernel<<<(N_NODES * IN_PAD + 255) / 256, 256, 0, stream>>>(x, x_pad);
    pack_weights_kernel<<<2, 256, 0, stream>>>(w1l, w1r, w2l, w2r, wpack);

    // 3) layer-1 edge scatter (+ degree)
    scatter1_kernel<<<(N_EDGES + 255) / 256, 256, 0, stream>>>(x, src, dst, agg1p, deg);

    // 4) layer-1 node GEMM via WMMA (12500 waves of 16 nodes, 4 waves/block)
    layer1_wmma_kernel<<<(N_NODES / 16 + 3) / 4, 128, 0, stream>>>(
        x_pad, agg1p, deg, wpack, b1, h1);

    // 5) layer-2 edge scatter (64 lanes per edge, coalesced)
    long long s2_threads = (long long)N_EDGES * 64;
    scatter2_kernel<<<(unsigned)(s2_threads / 256), 256, 0, stream>>>(h1, src, dst, agg2);

    // 6) layer-2 node GEMM via WMMA
    layer2_wmma_kernel<<<(N_NODES / 16 + 3) / 4, 128, 0, stream>>>(
        h1, agg2, deg, wpack, b2, h2);

    // 7) classifier head
    classifier_kernel<<<(N_NODES + 255) / 256, 256, 0, stream>>>(h2, wc, bc, out);
}
